// TriBlock_10376640987611
// MI455X (gfx1250) — matrix operand order repaired and checked by reading, not compiled, not run on hardware
//
#include <hip/hip_runtime.h>
#include <hip/hip_bf16.h>

// MI455X / gfx1250 triangle-attention block.
// C=128, CH=32, H=4, B=1, N=256. All matrix math: v_wmma_f32_16x16x32_f16.

#define NPOS 256
#define NN   65536   // 256*256
#define CDIM 128
#define NHEAD 4
#define CHD  32

typedef __attribute__((ext_vector_type(16))) _Float16 v16h;
typedef __attribute__((ext_vector_type(8)))  float    v8f;

__device__ __forceinline__ int lane_id() { return (int)(threadIdx.x & 31u); }

__device__ __forceinline__ v8f wmma_f16(v16h a, v16h b, v8f c) {
  return __builtin_amdgcn_wmma_f32_16x16x32_f16(false, a, false, b, (short)0, c, false, false);
}

__device__ __forceinline__ float fast_rcp(float x) { return __builtin_amdgcn_rcpf(x); }
__device__ __forceinline__ float fast_sigmoid(float x) {
  return fast_rcp(1.f + __expf(-x));
}

// ---- Fragment loaders (gfx1250 wave32; A and B both in contiguous k = 16*(lane/16)+e order, cf. CDNA5 ISA §7.12.2) ----

// A-matrix 16x32 f16 from row-major src (ld in halves).
__device__ __forceinline__ v16h load_a_frag(const _Float16* __restrict__ base, int ld) {
  int l = lane_id();
  int m = l & 15;
  int khi = (l & 16) ? 16 : 0;
  const _Float16* r0 = base + (size_t)m * ld + khi;
  const _Float16* r1 = r0 + 8;
  v16h a;
#pragma unroll
  for (int e = 0; e < 8; ++e) { a[e] = r0[e]; a[8 + e] = r1[e]; }
  return a;
}

// B-matrix 32x16, N-major storage: bt[n*ld + k].  B[k,n] = bt[n*ld+k].
__device__ __forceinline__ v16h load_b_frag_nmajor(const _Float16* __restrict__ bt, int ld) {
  int l = lane_id();
  int n = l & 15;
  int kb = (l & 16) ? 16 : 0;
  const _Float16* src = bt + (size_t)n * ld + kb;
  v16h b;
#pragma unroll
  for (int e = 0; e < 16; ++e) b[e] = src[e];
  return b;
}

// ------------------------------------------------------------------
// 0) cmap [C, N, N] f32 -> X [N*N, C] f32.  LDS-tiled transpose.
// ------------------------------------------------------------------
__global__ void transpose_in_kernel(const float* __restrict__ cmap, float* __restrict__ X) {
  __shared__ float tile[32][65];
  int t = (int)threadIdx.x;
  int p0 = blockIdx.x * 64;
  int c0 = blockIdx.y * 32;
#pragma unroll
  for (int r = 0; r < 8; ++r) {             // read: consecutive p (coalesced)
    int idx = r * 256 + t;
    int cl = idx >> 6, pl = idx & 63;
    tile[cl][pl] = cmap[(size_t)(c0 + cl) * NN + p0 + pl];
  }
  __syncthreads();
#pragma unroll
  for (int r = 0; r < 8; ++r) {             // write: consecutive c (coalesced)
    int idx = r * 256 + t;
    int pl = idx >> 5, cl = idx & 31;
    X[(size_t)(p0 + pl) * CDIM + c0 + cl] = tile[cl][pl];
  }
}

// ------------------------------------------------------------------
// 1) LayerNorm over C, fp32 -> f16.  One wave per position (4 ch/lane).
// ------------------------------------------------------------------
__global__ void ln_kernel(const float* __restrict__ src,
                          const float* __restrict__ w, const float* __restrict__ b,
                          _Float16* __restrict__ dst, int transpose) {
  int wave = (int)(threadIdx.x >> 5);
  int l = lane_id();
  int p = blockIdx.x * 8 + wave;
  int sp = transpose ? ((p & (NPOS - 1)) * NPOS + (p >> 8)) : p;
  const float* xr = src + (size_t)sp * CDIM;
  int c0 = l * 4;
  float v0[4];
#pragma unroll
  for (int k = 0; k < 4; ++k) v0[k] = xr[c0 + k];
  float s = 0.f, ss = 0.f;
#pragma unroll
  for (int k = 0; k < 4; ++k) { s += v0[k]; ss += v0[k] * v0[k]; }
#pragma unroll
  for (int m = 1; m < 32; m <<= 1) {
    s  += __shfl_xor(s,  m, 32);
    ss += __shfl_xor(ss, m, 32);
  }
  float mean = s * (1.f / CDIM);
  float var  = ss * (1.f / CDIM) - mean * mean;
  float ri = rsqrtf(var + 1e-5f);
  _Float16* dr = dst + (size_t)p * CDIM;
#pragma unroll
  for (int k = 0; k < 4; ++k)
    dr[c0 + k] = (_Float16)((v0[k] - mean) * ri * w[c0 + k] + b[c0 + k]);
}

// ------------------------------------------------------------------
// 2) WMMA GEMM: A[NN,128] f16 x W[128,128] f32 -> out[NN,128] f16
//    MODE 0: plain   MODE 1: * CH^-0.5 (q)   MODE 2: sigmoid(x+bias) (gate)
// ------------------------------------------------------------------
template <int MODE>
__global__ __launch_bounds__(256)
void gemm_proj_kernel(const _Float16* __restrict__ A,
                      const float* __restrict__ W,
                      const float* __restrict__ bias,
                      _Float16* __restrict__ out) {
  __shared__ _Float16 Wt[CDIM * CDIM];  // Wt[n*128 + k] = W[k*128 + n]
  int tid = (int)threadIdx.x;
  for (int i = tid; i < CDIM * CDIM; i += 256) {
    int k = i >> 7, n = i & (CDIM - 1);
    Wt[n * CDIM + k] = (_Float16)W[i];
  }
  __syncthreads();

  int wave = tid >> 5;
  int mrows = blockIdx.x * 128 + wave * 16;
  const _Float16* Ab = A + (size_t)mrows * CDIM;
  v16h a0 = load_a_frag(Ab +  0, CDIM);
  v16h a1 = load_a_frag(Ab + 32, CDIM);
  v16h a2 = load_a_frag(Ab + 64, CDIM);
  v16h a3 = load_a_frag(Ab + 96, CDIM);

  int l = lane_id();
  int ncol = l & 15;
  int mb = (l & 16) ? 8 : 0;

#pragma unroll
  for (int nt = 0; nt < 8; ++nt) {
    const _Float16* Bt = Wt + (size_t)nt * 16 * CDIM;
    v8f acc = {};
    acc = wmma_f16(a0, load_b_frag_nmajor(Bt +  0, CDIM), acc);
    acc = wmma_f16(a1, load_b_frag_nmajor(Bt + 32, CDIM), acc);
    acc = wmma_f16(a2, load_b_frag_nmajor(Bt + 64, CDIM), acc);
    acc = wmma_f16(a3, load_b_frag_nmajor(Bt + 96, CDIM), acc);
    int col = nt * 16 + ncol;
    float bv = (MODE == 2) ? bias[col] : 0.f;
#pragma unroll
    for (int r = 0; r < 8; ++r) {
      float v = acc[r];
      if (MODE == 1) v *= 0.17677669529663687f;          // 32^-0.5
      if (MODE == 2) v = fast_sigmoid(v + bv);
      out[(size_t)(mrows + mb + r) * CDIM + col] = (_Float16)v;
    }
  }
}

// ------------------------------------------------------------------
// 3) Pairwise bias, head-major out: TB[h*NN + p] = sum_c xln[p,c]*wz[c,h]
// ------------------------------------------------------------------
__global__ void zproj_kernel(const _Float16* __restrict__ X,
                             const float* __restrict__ wz, float* __restrict__ TB) {
  __shared__ float wzl[CDIM * NHEAD];
  int tid = (int)threadIdx.x;
#pragma unroll
  for (int i = tid; i < CDIM * NHEAD; i += 256) wzl[i] = wz[i];
  __syncthreads();

  int wave = tid >> 5;
  int l = lane_id();
  int p = blockIdx.x * 8 + wave;
  const _Float16* xr = X + (size_t)p * CDIM + l * 4;
  float acc[NHEAD] = {0.f, 0.f, 0.f, 0.f};
#pragma unroll
  for (int k = 0; k < 4; ++k) {
    float xv = (float)xr[k];
    int c = l * 4 + k;
#pragma unroll
    for (int h = 0; h < NHEAD; ++h) acc[h] += xv * wzl[c * NHEAD + h];
  }
#pragma unroll
  for (int h = 0; h < NHEAD; ++h)
#pragma unroll
    for (int m = 1; m < 32; m <<= 1) acc[h] += __shfl_xor(acc[h], m, 32);
  if (l == 0) {
#pragma unroll
    for (int h = 0; h < NHEAD; ++h) TB[(size_t)h * NN + p] = acc[h];
  }
}

// ------------------------------------------------------------------
// 4) Attention per (row i, head h), flash-style streaming softmax.
//    Block = 256 thr = 8 waves; wave owns 16 q-rows per task (2 tasks).
//    K processed in 2 chunks of 128 columns: 8 score WMMAs per chunk
//    (only 64 VGPRs of scores live), online max/sum rescale of the two
//    output accumulators, exp'd P staged in a 16x128 f16 LDS tile,
//    then 4x2 P@V WMMAs.  No scratch spills.
// ------------------------------------------------------------------
__global__ __launch_bounds__(256)
void attn_kernel(const _Float16* __restrict__ Q,
                 const _Float16* __restrict__ K,
                 const _Float16* __restrict__ V,
                 const _Float16* __restrict__ G,
                 const float* __restrict__ TB,
                 _Float16* __restrict__ O) {
  extern __shared__ __align__(16) _Float16 smem[];
  _Float16* kbuf = smem;                 // [256][32]  (j-major)
  _Float16* vt   = smem + NPOS * CHD;    // [32][256]  (d-major, transposed)
  int i = blockIdx.x, h = blockIdx.y;
  int tid = (int)threadIdx.x;

  const _Float16* Kg = K + ((size_t)i * NPOS) * CDIM + h * CHD;
  const _Float16* Vg = V + ((size_t)i * NPOS) * CDIM + h * CHD;
  {
    int j = tid;  // one row per thread
#pragma unroll
    for (int d = 0; d < CHD; ++d) {
      kbuf[j * CHD + d] = Kg[(size_t)j * CDIM + d];
      vt[d * NPOS + j] = Vg[(size_t)j * CDIM + d];
    }
  }
  __syncthreads();

  int wave = tid >> 5;
  int l = lane_id();
  int ncol = l & 15;
  int mb = (l & 16) ? 8 : 0;
  _Float16* pstage = smem + 2 * NPOS * CHD + (size_t)wave * 16 * 128;  // [16][128]
  const float* TBh = TB + (size_t)h * NN;

  for (int task = 0; task < 2; ++task) {
    int qt = wave * 2 + task;           // q-tile 0..15
    const _Float16* Qb = Q + ((size_t)(i * NPOS + qt * 16)) * CDIM + h * CHD;
    v16h aq = load_a_frag(Qb, CDIM);

    v8f oacc0 = {}, oacc1 = {};
    float rmax[8], rsum[8];
#pragma unroll
    for (int r = 0; r < 8; ++r) { rmax[r] = -1e30f; rsum[r] = 0.f; }

#pragma unroll
    for (int kb = 0; kb < 2; ++kb) {    // 128 columns per chunk
      // ---- scores for 8 k-tiles ----
      v8f s[8];
#pragma unroll
      for (int kt = 0; kt < 8; ++kt) {
        v8f z = {};
        s[kt] = wmma_f16(aq, load_b_frag_nmajor(kbuf + (kb * 8 + kt) * 16 * CHD, CHD), z);
      }
      // ---- pairwise bias (head-major, contiguous 64B per 16 lanes) ----
#pragma unroll
      for (int r = 0; r < 8; ++r) {
        int qrow = qt * 16 + mb + r;
        const float* tbr = TBh + (size_t)qrow * NPOS + kb * 128 + ncol;
#pragma unroll
        for (int kt = 0; kt < 8; ++kt) s[kt][r] += tbr[kt * 16];
      }
      // ---- online softmax update per row ----
#pragma unroll
      for (int r = 0; r < 8; ++r) {
        float cm = s[0][r];
#pragma unroll
        for (int kt = 1; kt < 8; ++kt) cm = fmaxf(cm, s[kt][r]);
#pragma unroll
        for (int m = 1; m < 16; m <<= 1) cm = fmaxf(cm, __shfl_xor(cm, m, 32));
        float nmax = fmaxf(rmax[r], cm);
        float scale = __expf(rmax[r] - nmax);
        rmax[r] = nmax;
        oacc0[r] *= scale;
        oacc1[r] *= scale;
        float csum = 0.f;
#pragma unroll
        for (int kt = 0; kt < 8; ++kt) {
          float e = __expf(s[kt][r] - nmax);
          s[kt][r] = e;
          csum += e;
        }
#pragma unroll
        for (int m = 1; m < 16; m <<= 1) csum += __shfl_xor(csum, m, 32);
        rsum[r] = rsum[r] * scale + csum;
        // stage exp'd P (f16) for this row
        _Float16* pr = pstage + (size_t)(mb + r) * 128 + ncol;
#pragma unroll
        for (int kt = 0; kt < 8; ++kt) pr[kt * 16] = (_Float16)s[kt][r];
      }
      // ---- accumulate O += P(16x128) @ V(128x32) ----
#pragma unroll
      for (int kc = 0; kc < 4; ++kc) {
        v16h ap = load_a_frag(pstage + kc * 32, 128);
        oacc0 = wmma_f16(ap, load_b_frag_nmajor(vt + kb * 128 + kc * 32, NPOS), oacc0);
        oacc1 = wmma_f16(ap, load_b_frag_nmajor(vt + 16 * NPOS + kb * 128 + kc * 32, NPOS), oacc1);
      }
    }
    // ---- normalize, gate, store ----
#pragma unroll
    for (int r = 0; r < 8; ++r) {
      float ri = fast_rcp(rsum[r]);
      size_t pos = (size_t)(i * NPOS + qt * 16 + mb + r);
      int c0 = h * CHD + ncol;
      float g0 = (float)G[pos * CDIM + c0];
      float g1 = (float)G[pos * CDIM + c0 + 16];
      O[pos * CDIM + c0]      = (_Float16)(oacc0[r] * ri * g0);
      O[pos * CDIM + c0 + 16] = (_Float16)(oacc1[r] * ri * g1);
    }
  }
}

// ------------------------------------------------------------------
// 5) Output projection.  PASS 0: X[p,c] += O@wo + bo (residual, in place).
//    PASS 1: Y[p,c] = O@wo + bo  (dense; transpose handled by final kernel).
// ------------------------------------------------------------------
template <int PASS>
__global__ __launch_bounds__(256)
void out_proj_kernel(const _Float16* __restrict__ Ob,
                     const float* __restrict__ wo,
                     const float* __restrict__ bo,
                     float* __restrict__ X,
                     float* __restrict__ Y) {
  __shared__ _Float16 Wt[CDIM * CDIM];
  __shared__ float bol[CDIM];
  int tid = (int)threadIdx.x;
  for (int i = tid; i < CDIM * CDIM; i += 256) {
    int k = i >> 7, n = i & (CDIM - 1);
    Wt[n * CDIM + k] = (_Float16)wo[i];
  }
  if (tid < CDIM) bol[tid] = bo[tid];
  __syncthreads();

  int wave = tid >> 5;
  int mrows = blockIdx.x * 128 + wave * 16;
  const _Float16* Ab = Ob + (size_t)mrows * CDIM;
  v16h a0 = load_a_frag(Ab +  0, CDIM);
  v16h a1 = load_a_frag(Ab + 32, CDIM);
  v16h a2 = load_a_frag(Ab + 64, CDIM);
  v16h a3 = load_a_frag(Ab + 96, CDIM);

  int l = lane_id();
  int ncol = l & 15;
  int mb = (l & 16) ? 8 : 0;

#pragma unroll
  for (int nt = 0; nt < 8; ++nt) {
    const _Float16* Bt = Wt + (size_t)nt * 16 * CDIM;
    v8f acc = {};
    acc = wmma_f16(a0, load_b_frag_nmajor(Bt +  0, CDIM), acc);
    acc = wmma_f16(a1, load_b_frag_nmajor(Bt + 32, CDIM), acc);
    acc = wmma_f16(a2, load_b_frag_nmajor(Bt + 64, CDIM), acc);
    acc = wmma_f16(a3, load_b_frag_nmajor(Bt + 96, CDIM), acc);
    int col = nt * 16 + ncol;
#pragma unroll
    for (int r = 0; r < 8; ++r) {
      size_t row = (size_t)(mrows + mb + r);
      float y = acc[r] + bol[col];
      if (PASS == 0) X[row * CDIM + col] += y;
      else           Y[row * CDIM + col] = y;
    }
  }
}

// ------------------------------------------------------------------
// 6) Final: out[c, i, j] = X[i,j,c] + Y[j,i,c].  LDS-tiled output transpose.
// ------------------------------------------------------------------
__global__ void transpose_out_kernel(const float* __restrict__ X,
                                     const float* __restrict__ Y,
                                     float* __restrict__ out) {
  __shared__ float tile[32][65];
  int t = (int)threadIdx.x;
  int p0 = blockIdx.x * 64;               // p = i*256 + j; one i per block
  int c0 = blockIdx.y * 32;
  int i = p0 >> 8;
  int j0 = p0 & (NPOS - 1);
#pragma unroll
  for (int r = 0; r < 8; ++r) {           // read: consecutive c (coalesced)
    int idx = r * 256 + t;
    int pl = idx >> 5, cl = idx & 31;
    int j = j0 + pl;
    float xv = X[((size_t)(i * NPOS + j)) * CDIM + c0 + cl];
    float yv = Y[((size_t)(j * NPOS + i)) * CDIM + c0 + cl];
    tile[cl][pl] = xv + yv;
  }
  __syncthreads();
#pragma unroll
  for (int r = 0; r < 8; ++r) {           // write: consecutive j (coalesced)
    int idx = r * 256 + t;
    int cl = idx >> 6, pl = idx & 63;
    out[(size_t)(c0 + cl) * NN + p0 + pl] = tile[cl][pl];
  }
}

// ------------------------------------------------------------------
extern "C" void kernel_launch(void* const* d_in, const int* in_sizes, int n_in,
                              void* d_out, int out_size, void* d_ws, size_t ws_size,
                              hipStream_t stream) {
  (void)in_sizes; (void)n_in; (void)out_size; (void)ws_size;
  const float* cmap = (const float*)d_in[0];

  // workspace carve-up (~169 MB)
  char* ws = (char*)d_ws;
  float*     X    = (float*)ws;        ws += (size_t)NN * CDIM * sizeof(float);
  float*     Y    = (float*)ws;        ws += (size_t)NN * CDIM * sizeof(float);
  _Float16*  xln  = (_Float16*)ws;     ws += (size_t)NN * CDIM * sizeof(_Float16);
  _Float16*  qb   = (_Float16*)ws;     ws += (size_t)NN * CDIM * sizeof(_Float16);
  _Float16*  kb   = (_Float16*)ws;     ws += (size_t)NN * CDIM * sizeof(_Float16);
  _Float16*  vb   = (_Float16*)ws;     ws += (size_t)NN * CDIM * sizeof(_Float16);
  _Float16*  gb   = (_Float16*)ws;     ws += (size_t)NN * CDIM * sizeof(_Float16);
  float*     tbb  = (float*)ws;        ws += (size_t)NN * NHEAD * sizeof(float);
  _Float16*  ob   = (_Float16*)ws;

  transpose_in_kernel<<<dim3(NN / 64, CDIM / 32), 256, 0, stream>>>(cmap, X);

  // kbuf 16K + vt 16K + 8 waves * 4K pstage = 64 KB dynamic LDS
  const size_t attn_lds = (size_t)(2 * NPOS * CHD + 8 * 16 * 128) * sizeof(_Float16);

  for (int pass = 0; pass < 2; ++pass) {
    const float* ln_w = (const float*)d_in[1 + pass * 10 + 0];
    const float* ln_b = (const float*)d_in[1 + pass * 10 + 1];
    const float* wq   = (const float*)d_in[1 + pass * 10 + 2];
    const float* wk   = (const float*)d_in[1 + pass * 10 + 3];
    const float* wv   = (const float*)d_in[1 + pass * 10 + 4];
    const float* wz   = (const float*)d_in[1 + pass * 10 + 5];
    const float* wg   = (const float*)d_in[1 + pass * 10 + 6];
    const float* bg   = (const float*)d_in[1 + pass * 10 + 7];
    const float* wo   = (const float*)d_in[1 + pass * 10 + 8];
    const float* bo   = (const float*)d_in[1 + pass * 10 + 9];

    ln_kernel<<<NN / 8, 256, 0, stream>>>(X, ln_w, ln_b, xln, pass);

    gemm_proj_kernel<1><<<NN / 128, 256, 0, stream>>>(xln, wq, nullptr, qb);
    gemm_proj_kernel<0><<<NN / 128, 256, 0, stream>>>(xln, wk, nullptr, kb);
    gemm_proj_kernel<0><<<NN / 128, 256, 0, stream>>>(xln, wv, nullptr, vb);
    gemm_proj_kernel<2><<<NN / 128, 256, 0, stream>>>(xln, wg, bg,      gb);
    zproj_kernel<<<NN / 8, 256, 0, stream>>>(xln, wz, tbb);

    attn_kernel<<<dim3(NPOS, NHEAD), 256, attn_lds, stream>>>(qb, kb, vb, gb, tbb, ob);

    if (pass == 0)
      out_proj_kernel<0><<<NN / 128, 256, 0, stream>>>(ob, wo, bo, X, Y);
    else
      out_proj_kernel<1><<<NN / 128, 256, 0, stream>>>(ob, wo, bo, X, Y);
  }

  transpose_out_kernel<<<dim3(NN / 64, CDIM / 32), 256, 0, stream>>>(X, Y, (float*)d_out);
}
